// FlexAttention_77575699300897
// MI455X (gfx1250) — matrix operand order repaired and checked by reading, not compiled, not run on hardware
//
#include <hip/hip_runtime.h>

// ---------------------------------------------------------------------------
// Sliding-window causal flash attention for MI455X (gfx1250, wave32, WMMA).
// B=2 H=16 L=2048 E=128 window=512, fp32 in/out, bf16 WMMA math, fp32 softmax.
//
// Memory-bound problem (~128 MiB @ 23.3 TB/s ~= 5.8 us floor); bf16 WMMA gives
// orders-of-magnitude compute headroom, so the kernel is organized around
// streaming K/V once per query block through double-buffered LDS.
// ---------------------------------------------------------------------------

typedef __attribute__((ext_vector_type(16))) __bf16 bf16x16;
typedef __attribute__((ext_vector_type(8)))  __bf16 bf16x8;
typedef __attribute__((ext_vector_type(4)))  __bf16 bf16x4;
typedef __attribute__((ext_vector_type(8)))  float  f32x8;
typedef __attribute__((ext_vector_type(4)))  float  f32x4;

namespace {
constexpr int NB   = 2;     // batch
constexpr int NH   = 16;    // heads
constexpr int SEQ  = 2048;  // L
constexpr int DH   = 128;   // E
constexpr int WIN  = 512;   // sliding window
constexpr int BM   = 64;    // query rows per workgroup (16 per wave)
constexpr int BN   = 32;    // keys per inner block
constexpr int NWAVES   = 4;
constexpr int NTHREADS = 128;
// LDS strides (bf16 elements), padded for 16B alignment + bank spread
constexpr int KSTR = DH + 8;  // 136 -> 272B rows (16B aligned, conflict-free)
constexpr int VSTR = BN + 8;  // 40  -> 80B rows
constexpr int PSTR = BN + 8;  // 40
}

__device__ __forceinline__ f32x8 wmma_bf16f32(bf16x16 a, bf16x16 b, f32x8 c) {
  // D = A(16x32 bf16) * B(32x16 bf16) + C(16x16 f32)
  return __builtin_amdgcn_wmma_f32_16x16x32_bf16(
      /*neg_a=*/false, a, /*neg_b=*/false, b,
      /*c_mod=*/(short)0, c, /*reuse_a=*/false, /*reuse_b=*/false);
}

__device__ __forceinline__ bf16x16 cat8(bf16x8 lo, bf16x8 hi) {
  return __builtin_shufflevector(lo, hi, 0,1,2,3,4,5,6,7,8,9,10,11,12,13,14,15);
}

__global__ __launch_bounds__(NTHREADS) void fa_swa_wmma_kernel(
    const float* __restrict__ qg_, const float* __restrict__ kg_,
    const float* __restrict__ vg_, float* __restrict__ og_) {
  // K tile:  [key][e]   (bf16, row stride 136 -> ds_load_b128 fragments)
  // Vt tile: [e][key]   (bf16, transposed during staging)
  // P tile:  per-wave P^T relayout buffer [q][key]
  __shared__ __bf16 Kls [2][BN][KSTR];
  __shared__ __bf16 Vtls[2][DH][VSTR];
  __shared__ __bf16 Pls [NWAVES][16][PSTR];

  const int tid  = threadIdx.x;
  const int lane = tid & 31;
  const int wave = tid >> 5;
  const int hh   = lane >> 4;   // lane half: selects M-offset 8 / K-offset 8
  const int qn   = lane & 15;   // query column owned by this lane

  const int    qb    = blockIdx.x * BM;
  const size_t bhoff = (size_t)blockIdx.y * SEQ * DH;
  const float* qg = qg_ + bhoff;
  const float* kg = kg_ + bhoff;
  const float* vg = vg_ + bhoff;
  float*       og = og_ + bhoff;

  const int qw   = qb + wave * 16;  // wave's first query row
  const int qidx = qw + qn;         // this lane's query row
  // fold softmax scale and log2(e) into Q so p = exp2(s - m)
  constexpr float SC = 0.08838834764831845f * 1.4426950408889634f;

  // ---- Q^T B-fragments, 4 chunks of 32 e, prescaled, kept in registers ----
  bf16x16 qf[4];
  {
    const float* qrow = qg + (size_t)qidx * DH;
    #pragma unroll
    for (int c = 0; c < 4; ++c) {
      const int eb = 32 * c + 8 * hh;    // B-frag: e runs [8h,8h+7] & [16+8h,+7]
      f32x4 t0 = *(const f32x4*)(qrow + eb);
      f32x4 t1 = *(const f32x4*)(qrow + eb + 4);
      f32x4 t2 = *(const f32x4*)(qrow + eb + 16);
      f32x4 t3 = *(const f32x4*)(qrow + eb + 20);
      #pragma unroll
      for (int j = 0; j < 4; ++j) {
        qf[c][j]      = (__bf16)(t0[j] * SC);
        qf[c][4 + j]  = (__bf16)(t1[j] * SC);
        qf[c][8 + j]  = (__bf16)(t2[j] * SC);
        qf[c][12 + j] = (__bf16)(t3[j] * SC);
      }
    }
  }

  const f32x8 vzero = {0.f,0.f,0.f,0.f,0.f,0.f,0.f,0.f};
  f32x8 o[8];                       // O^T accumulators: 8 tiles of (16e x 16q)
  #pragma unroll
  for (int i = 0; i < 8; ++i) o[i] = vzero;
  float mrow = -1e30f, lrow = 0.f;  // online softmax stats (per lane = per query)

  int kb0 = qb - WIN; if (kb0 < 0) kb0 = 0;
  const int nblk = (qb + BM - kb0) / BN;     // <= 18 blocks of 32 keys

  // staging assignment: thread covers 4 consecutive e at 8 key rows
  const int se0 = (tid & 31) << 2;  // e: 0..124 step 4
  const int skr = tid >> 5;         // key group 0..3

  // ---- stage block 0 into buffer 0 (fp32->bf16, V transposed) ----
  #pragma unroll
  for (int i = 0; i < 8; ++i) {
    const int key = skr * 8 + i;
    f32x4 kv = *(const f32x4*)(kg + (size_t)(kb0 + key) * DH + se0);
    f32x4 vv = *(const f32x4*)(vg + (size_t)(kb0 + key) * DH + se0);
    bf16x4 kc; kc[0]=(__bf16)kv[0]; kc[1]=(__bf16)kv[1]; kc[2]=(__bf16)kv[2]; kc[3]=(__bf16)kv[3];
    *(bf16x4*)&Kls[0][key][se0] = kc;
    Vtls[0][se0 + 0][key] = (__bf16)vv[0];
    Vtls[0][se0 + 1][key] = (__bf16)vv[1];
    Vtls[0][se0 + 2][key] = (__bf16)vv[2];
    Vtls[0][se0 + 3][key] = (__bf16)vv[3];
  }
  __syncthreads();

  for (int it = 0; it < nblk; ++it) {
    const int buf = it & 1;
    const int kb  = kb0 + it * BN;

    // ---- prefetch next key block into the other buffer (overlaps compute) ----
    if (it + 1 < nblk) {
      const int kbn = kb + BN;
      #pragma unroll
      for (int i = 0; i < 8; ++i) {
        const int key = skr * 8 + i;
        f32x4 kv = *(const f32x4*)(kg + (size_t)(kbn + key) * DH + se0);
        f32x4 vv = *(const f32x4*)(vg + (size_t)(kbn + key) * DH + se0);
        bf16x4 kc; kc[0]=(__bf16)kv[0]; kc[1]=(__bf16)kv[1]; kc[2]=(__bf16)kv[2]; kc[3]=(__bf16)kv[3];
        *(bf16x4*)&Kls[buf ^ 1][key][se0] = kc;
        Vtls[buf ^ 1][se0 + 0][key] = (__bf16)vv[0];
        Vtls[buf ^ 1][se0 + 1][key] = (__bf16)vv[1];
        Vtls[buf ^ 1][se0 + 2][key] = (__bf16)vv[2];
        Vtls[buf ^ 1][se0 + 3][key] = (__bf16)vv[3];
      }
    }

    // wave-uniform skip of blocks fully outside this wave's window
    // (EXEC stays all-ones inside -> WMMA legal)
    const bool useful = (kb <= qw + 15) && (kb + BN - 1 >= qw - WIN);
    if (useful) {
      // ---- S^T = K * Q^T : two 16-key subtiles, K-dim = 128 e in 4 chunks ----
      f32x8 s0 = vzero, s1 = vzero;
      const __bf16* kr0 = &Kls[buf][qn][0];       // A row M = key (subtile 0)
      const __bf16* kr1 = &Kls[buf][16 + qn][0];  // subtile 1
      #pragma unroll
      for (int c = 0; c < 4; ++c) {
        const int ea = 32 * c + 8 * hh;  // A-frag: e runs [8h,8h+7] & [16+8h,+7]
        bf16x16 a0 = cat8(*(const bf16x8*)(kr0 + ea), *(const bf16x8*)(kr0 + ea + 16));
        bf16x16 a1 = cat8(*(const bf16x8*)(kr1 + ea), *(const bf16x8*)(kr1 + ea + 16));
        s0 = wmma_bf16f32(a0, qf[c], s0);
        s1 = wmma_bf16f32(a1, qf[c], s1);
      }

      // ---- mask: causal + window, only on boundary blocks ----
      // interior (wave-uniform): every (q,k) pair valid -> skip all mask ALU
      const bool interior = (kb + BN - 1 <= qw) && (kb >= qw + 15 - WIN);
      if (!interior) {
        // valid  <=>  0 <= qidx-k <= WIN  <=>  (unsigned)(qidx-k) <= WIN
        const int d0 = qidx - (kb + 8 * hh);       // subtile0: k = kb+8h+r
        const int d1 = d0 - 16;                    // subtile1: k = kb+16+8h+r
        #pragma unroll
        for (int r = 0; r < 8; ++r) {
          s0[r] = ((unsigned)(d0 - r) <= (unsigned)WIN) ? s0[r] : -1e30f;
          s1[r] = ((unsigned)(d1 - r) <= (unsigned)WIN) ? s1[r] : -1e30f;
        }
      }

      // ---- online softmax: column stats = per-lane + one xor-16 shuffle ----
      float tmax = -1e30f;
      #pragma unroll
      for (int r = 0; r < 8; ++r) tmax = fmaxf(tmax, fmaxf(s0[r], s1[r]));
      tmax = fmaxf(tmax, __shfl_xor(tmax, 16));
      const float mnew  = fmaxf(mrow, tmax);
      const float alpha = exp2f(mrow - mnew);
      mrow = mnew;
      float psum = 0.f;
      #pragma unroll
      for (int r = 0; r < 8; ++r) {
        s0[r] = exp2f(s0[r] - mnew);
        s1[r] = exp2f(s1[r] - mnew);
        psum += s0[r] + s1[r];
      }
      psum += __shfl_xor(psum, 16);
      lrow = lrow * alpha + psum;
      #pragma unroll
      for (int i = 0; i < 8; ++i) o[i] *= alpha;   // per-lane column rescale

      // ---- P^T relayout via LDS: C-layout (key split by 8) -> B-frag (also by 8) ----
      bf16x8 p0, p1;
      #pragma unroll
      for (int r = 0; r < 8; ++r) { p0[r] = (__bf16)s0[r]; p1[r] = (__bf16)s1[r]; }
      *(bf16x8*)&Pls[wave][qn][8 * hh]      = p0;
      *(bf16x8*)&Pls[wave][qn][16 + 8 * hh] = p1;
      const bf16x16 pb = cat8(*(const bf16x8*)&Pls[wave][qn][8 * hh],
                              *(const bf16x8*)&Pls[wave][qn][16 + 8 * hh]);

      // ---- O^T += V^T * P^T over all 8 e-chunks ----
      #pragma unroll
      for (int ec = 0; ec < 8; ++ec) {
        const __bf16* vr = &Vtls[buf][16 * ec + qn][0];  // A row M = e_local
        bf16x16 va = cat8(*(const bf16x8*)(vr + 8 * hh),
                          *(const bf16x8*)(vr + 16 + 8 * hh));
        o[ec] = wmma_bf16f32(va, pb, o[ec]);
      }
    }
    __syncthreads();  // next iteration reads the buffer staged above
  }

  // ---- epilogue: out[q][e] = O^T[e][q] / l[q]  (vectorized f32x4 stores) ----
  const float inv = 1.0f / lrow;
  float* orow = og + (size_t)qidx * DH;
  #pragma unroll
  for (int ec = 0; ec < 8; ++ec) {
    f32x4 lo = { o[ec][0]*inv, o[ec][1]*inv, o[ec][2]*inv, o[ec][3]*inv };
    f32x4 hi = { o[ec][4]*inv, o[ec][5]*inv, o[ec][6]*inv, o[ec][7]*inv };
    *(f32x4*)(orow + 16 * ec + 8 * hh)     = lo;   // e = 16*ec + 8h + r
    *(f32x4*)(orow + 16 * ec + 8 * hh + 4) = hi;
  }
}

extern "C" void kernel_launch(void* const* d_in, const int* in_sizes, int n_in,
                              void* d_out, int out_size, void* d_ws, size_t ws_size,
                              hipStream_t stream) {
  (void)in_sizes; (void)n_in; (void)out_size; (void)d_ws; (void)ws_size;
  const float* q = (const float*)d_in[0];
  const float* k = (const float*)d_in[1];
  const float* v = (const float*)d_in[2];
  float* out = (float*)d_out;
  dim3 grid(SEQ / BM, NB * NH);   // 32 x 32 workgroups, 4 wave32s each
  fa_swa_wmma_kernel<<<grid, NTHREADS, 0, stream>>>(q, k, v, out);
}
